// GNNBlock_824633721540
// MI455X (gfx1250) — matrix operand order repaired and checked by reading, not compiled, not run on hardware
//
#include <hip/hip_runtime.h>

typedef __attribute__((ext_vector_type(16))) _Float16 v16h;
typedef __attribute__((ext_vector_type(4)))  _Float16 v4h;
typedef __attribute__((ext_vector_type(8)))  float    v8f;

#define D 128
#define LN_EPS 1e-5f

// ---------- helpers ----------
__device__ __forceinline__ v16h ld_frag2(const _Float16* p0, const _Float16* p1) {
    union { v16h v; struct { float4 a; float4 b; } f; } u;
    u.f.a = *(const float4*)p0;
    u.f.b = *(const float4*)p1;
    return u.v;
}

// ---------- 1. convert f32 weights -> f16 (row-major preserved) ----------
__global__ void cvt_w_kernel(const float* __restrict__ Wp,
                             const float* __restrict__ Wl,
                             const float* __restrict__ Wr,
                             _Float16* __restrict__ Wp16,
                             _Float16* __restrict__ Wl16,
                             _Float16* __restrict__ Wr16) {
    int t = blockIdx.x * blockDim.x + threadIdx.x;  // 0 .. 3*16384-1
    int which = t >> 14;
    int idx   = t & 16383;
    if (which == 0)      Wp16[idx] = (_Float16)Wp[idx];
    else if (which == 1) Wl16[idx] = (_Float16)Wl[idx];
    else                 Wr16[idx] = (_Float16)Wr[idx];
}

// ---------- 2. fused LayerNorm + ReLU + dropout -> h (f16), one wave/row ----
__global__ void ln_kernel(const float* __restrict__ x,
                          const float* __restrict__ mask,
                          const float* __restrict__ gamma,
                          const float* __restrict__ beta,
                          _Float16* __restrict__ h, int N) {
    int wave = (blockIdx.x * blockDim.x + threadIdx.x) >> 5;
    int lane = threadIdx.x & 31;
    if (wave >= N) return;
    const float4 xv = *(const float4*)(x + (size_t)wave * D + lane * 4);
    float s  = xv.x + xv.y + xv.z + xv.w;
    float ss = xv.x*xv.x + xv.y*xv.y + xv.z*xv.z + xv.w*xv.w;
    #pragma unroll
    for (int m = 16; m > 0; m >>= 1) {
        s  += __shfl_xor(s,  m, 32);
        ss += __shfl_xor(ss, m, 32);
    }
    float mu  = s * (1.0f / D);
    float var = ss * (1.0f / D) - mu * mu;
    float inv = rsqrtf(var + LN_EPS);
    float4 g  = *(const float4*)(gamma + lane * 4);
    float4 b  = *(const float4*)(beta  + lane * 4);
    float4 mk = *(const float4*)(mask + (size_t)wave * D + lane * 4);
    v4h o;
    o.x = (_Float16)(fmaxf((xv.x - mu) * inv * g.x + b.x, 0.0f) * mk.x);
    o.y = (_Float16)(fmaxf((xv.y - mu) * inv * g.y + b.y, 0.0f) * mk.y);
    o.z = (_Float16)(fmaxf((xv.z - mu) * inv * g.z + b.z, 0.0f) * mk.z);
    o.w = (_Float16)(fmaxf((xv.w - mu) * inv * g.w + b.w, 0.0f) * mk.w);
    *(v4h*)(h + (size_t)wave * D + lane * 4) = o;
}

// ---------- 3. p = relu(h @ Wp^T + bp), WMMA f16, one 16x16 tile / wave -----
template <bool GUARD>
__global__ void gemm_p_kernel(const _Float16* __restrict__ h,
                              const _Float16* __restrict__ W,
                              const float* __restrict__ bias,
                              float* __restrict__ p, int N) {
    int lane = threadIdx.x & 31;
    int wv   = threadIdx.x >> 5;             // 0..7 -> column tile
    int mbase = blockIdx.x * 16;
    int nbase = wv * 16;
    int r = lane & 15, half = lane >> 4;
    int arow = mbase + r;
    if (GUARD && arow >= N) arow = N - 1;
    float bv = bias[nbase + r];
    v8f acc = {};
    #pragma unroll
    for (int k0 = 0; k0 < D; k0 += 32) {
        const _Float16* pa = h + (size_t)arow * D + k0 + 8 * half;
        v16h a = ld_frag2(pa, pa + 16);
        const _Float16* pb = W + (size_t)(nbase + r) * D + k0 + 8 * half;
        v16h b = ld_frag2(pb, pb + 16);
        acc = __builtin_amdgcn_wmma_f32_16x16x32_f16(false, a, false, b,
                                                     (short)0, acc, false, false);
    }
    // lane owns column nbase+r, rows mbase + v + 8*half
    float* po = p + (size_t)(mbase + 8 * half) * D + nbase + r;
    #pragma unroll
    for (int v = 0; v < 8; ++v) {
        if (!GUARD || (mbase + v + 8 * half) < N)
            po[(size_t)v * D] = fmaxf(acc[v] + bv, 0.0f);
    }
}

// ---------- 4. edge scatter: one wave per edge, float4 per lane -------------
__global__ void scatter_kernel(const float* __restrict__ p,
                               const int* __restrict__ src,
                               const int* __restrict__ tgt,
                               float* __restrict__ sums,
                               float* __restrict__ cnt, int E) {
    long long t = (long long)blockIdx.x * blockDim.x + threadIdx.x;
    int e    = (int)(t >> 5);
    int lane = (int)(t & 31);
    if (e >= E) return;
    int s = src[e];
    int d = tgt[e];
    const float4 v = *(const float4*)(p + (size_t)s * D + lane * 4);
    float* o = sums + (size_t)d * D + lane * 4;
    atomicAdd(o + 0, v.x);
    atomicAdd(o + 1, v.y);
    atomicAdd(o + 2, v.z);
    atomicAdd(o + 3, v.w);
    if (lane == 0) atomicAdd(cnt + d, 1.0f);
}

// ---------- 5. agg = sums / max(cnt,1) -> f16 -------------------------------
__global__ void agg_kernel(const float* __restrict__ sums,
                           const float* __restrict__ cnt,
                           _Float16* __restrict__ agg, int N) {
    int t = blockIdx.x * blockDim.x + threadIdx.x;
    if (t >= N * (D / 4)) return;
    int row = t >> 5;          // D/4 == 32
    int c4  = (t & 31) * 4;
    float inv = 1.0f / fmaxf(cnt[row], 1.0f);
    float4 s = *(const float4*)(sums + (size_t)row * D + c4);
    v4h o;
    o.x = (_Float16)(s.x * inv);
    o.y = (_Float16)(s.y * inv);
    o.z = (_Float16)(s.z * inv);
    o.w = (_Float16)(s.w * inv);
    *(v4h*)(agg + (size_t)row * D + c4) = o;
}

// ---------- 6. out = agg @ Wl^T + bl + h @ Wr^T, dual-WMMA fused ------------
template <bool GUARD>
__global__ void gemm_out_kernel(const _Float16* __restrict__ agg,
                                const _Float16* __restrict__ h,
                                const _Float16* __restrict__ Wl,
                                const _Float16* __restrict__ Wr,
                                const float* __restrict__ bl,
                                float* __restrict__ out, int N) {
    int lane = threadIdx.x & 31;
    int wv   = threadIdx.x >> 5;
    int mbase = blockIdx.x * 16;
    int nbase = wv * 16;
    int r = lane & 15, half = lane >> 4;
    int arow = mbase + r;
    if (GUARD && arow >= N) arow = N - 1;
    float bv = bl[nbase + r];
    v8f acc = {};
    #pragma unroll
    for (int k0 = 0; k0 < D; k0 += 32) {
        const _Float16* pa1 = agg + (size_t)arow * D + k0 + 8 * half;
        v16h a1 = ld_frag2(pa1, pa1 + 16);
        const _Float16* pb1 = Wl + (size_t)(nbase + r) * D + k0 + 8 * half;
        v16h b1 = ld_frag2(pb1, pb1 + 16);
        acc = __builtin_amdgcn_wmma_f32_16x16x32_f16(false, a1, false, b1,
                                                     (short)0, acc, false, false);
        const _Float16* pa2 = h + (size_t)arow * D + k0 + 8 * half;
        v16h a2 = ld_frag2(pa2, pa2 + 16);
        const _Float16* pb2 = Wr + (size_t)(nbase + r) * D + k0 + 8 * half;
        v16h b2 = ld_frag2(pb2, pb2 + 16);
        acc = __builtin_amdgcn_wmma_f32_16x16x32_f16(false, a2, false, b2,
                                                     (short)0, acc, false, false);
    }
    float* po = out + (size_t)(mbase + 8 * half) * D + nbase + r;
    #pragma unroll
    for (int v = 0; v < 8; ++v) {
        if (!GUARD || (mbase + v + 8 * half) < N)
            po[(size_t)v * D] = acc[v] + bv;
    }
}

extern "C" void kernel_launch(void* const* d_in, const int* in_sizes, int n_in,
                              void* d_out, int out_size, void* d_ws, size_t ws_size,
                              hipStream_t stream) {
    const float* x     = (const float*)d_in[0];
    const float* mask  = (const float*)d_in[1];
    const float* gamma = (const float*)d_in[2];
    const float* beta  = (const float*)d_in[3];
    const float* Wp    = (const float*)d_in[4];
    const float* bp    = (const float*)d_in[5];
    const float* Wl    = (const float*)d_in[6];
    const float* bl    = (const float*)d_in[7];
    const float* Wr    = (const float*)d_in[8];
    const int*   eidx  = (const int*)d_in[9];

    const int N = in_sizes[0] / D;        // 50000
    const int E = in_sizes[9] / 2;        // 800000
    const int* src = eidx;
    const int* tgt = eidx + E;

    // ---- workspace layout (256B aligned) ----
    char* ws = (char*)d_ws;
    size_t off = 0;
    auto take = [&](size_t bytes) { size_t o = off; off = (off + bytes + 255) & ~(size_t)255; return o; };
    size_t o_h    = take((size_t)N * D * sizeof(_Float16));
    size_t o_agg  = take((size_t)N * D * sizeof(_Float16));
    size_t o_p    = take((size_t)N * D * sizeof(float));
    size_t o_sums = take((size_t)N * D * sizeof(float));
    size_t o_cnt  = take((size_t)N * sizeof(float));
    size_t o_wp   = take((size_t)D * D * sizeof(_Float16));
    size_t o_wl   = take((size_t)D * D * sizeof(_Float16));
    size_t o_wr   = take((size_t)D * D * sizeof(_Float16));

    _Float16* h16   = (_Float16*)(ws + o_h);
    _Float16* agg16 = (_Float16*)(ws + o_agg);
    float*    p32   = (float*)(ws + o_p);
    float*    sums  = (float*)(ws + o_sums);
    float*    cnt   = (float*)(ws + o_cnt);
    _Float16* Wp16  = (_Float16*)(ws + o_wp);
    _Float16* Wl16  = (_Float16*)(ws + o_wl);
    _Float16* Wr16  = (_Float16*)(ws + o_wr);

    const bool exact = (N % 16) == 0;
    const int mblocks = (N + 15) / 16;

    // 1. weights -> f16
    cvt_w_kernel<<<(3 * D * D + 255) / 256, 256, 0, stream>>>(Wp, Wl, Wr, Wp16, Wl16, Wr16);

    // 2. LN + ReLU + dropout (one wave per row)
    ln_kernel<<<(N * 32 + 255) / 256, 256, 0, stream>>>(x, mask, gamma, beta, h16, N);

    // 3. p = relu(h @ Wp^T + bp)
    if (exact)
        gemm_p_kernel<false><<<mblocks, 256, 0, stream>>>(h16, Wp16, bp, p32, N);
    else
        gemm_p_kernel<true><<<mblocks, 256, 0, stream>>>(h16, Wp16, bp, p32, N);

    // 4. zero accumulators (sums + cnt contiguous region)
    hipMemsetAsync(ws + o_sums, 0, (o_cnt - o_sums) + (size_t)N * sizeof(float), stream);

    // 5. edge scatter (one wave per edge, 512B contiguous per edge)
    long long sthreads = (long long)E * 32;
    scatter_kernel<<<(unsigned)((sthreads + 255) / 256), 256, 0, stream>>>(p32, src, tgt, sums, cnt, E);

    // 6. mean aggregate -> f16
    agg_kernel<<<(N * (D / 4) + 255) / 256, 256, 0, stream>>>(sums, cnt, agg16, N);

    // 7. out = agg @ Wl^T + bl + h @ Wr^T
    if (exact)
        gemm_out_kernel<false><<<mblocks, 256, 0, stream>>>(agg16, h16, Wl16, Wr16, bl, (float*)d_out, N);
    else
        gemm_out_kernel<true><<<mblocks, 256, 0, stream>>>(agg16, h16, Wl16, Wr16, bl, (float*)d_out, N);
}
